// SparseMultiheadSASphereConcat_42717744726200
// MI455X (gfx1250) — compile-verified, matrix operand order checked by reading, not run on hardware
//
#include <hip/hip_runtime.h>
#include <hip/hip_bf16.h>
#include <math.h>

#define N_PTS 20000
#define KNBR  16
#define M_EDG (N_PTS * KNBR)
#define CCH   384
#define HD_   64
#define H1_   3
#define A_CONST 0.0125f   // 0.05 * 0.25
#define PI_F 3.14159265358979323846f

typedef __attribute__((ext_vector_type(16))) __bf16 v16bf;
typedef __attribute__((ext_vector_type(8)))  float  v8f;
typedef unsigned short u16;

union FragAB { unsigned u[8]; v16bf v; };

__device__ __forceinline__ u16 f2bf(float f) {
  unsigned u = __builtin_bit_cast(unsigned, f);
  u += 0x7fffu + ((u >> 16) & 1u);           // round-to-nearest-even
  return (u16)(u >> 16);
}

// ---------------------------------------------------------------------------
// One-time fp32 -> bf16 conversion (weights / activations), 2 elems per thread
// ---------------------------------------------------------------------------
__global__ __launch_bounds__(256)
void f32_to_bf16_kernel(const float* __restrict__ src,
                        unsigned* __restrict__ dst, int npairs)
{
  const int i = blockIdx.x * blockDim.x + threadIdx.x;
  if (i >= npairs) return;
  const float a = src[2 * i], b = src[2 * i + 1];
  dst[i] = (unsigned)f2bf(a) | ((unsigned)f2bf(b) << 16);
}

// ---------------------------------------------------------------------------
// Tiled bf16 WMMA GEMM:  C[M,N] = A[M,K] @ B[K,N] + bias ; cols < scaleCols
// get multiplied by scaleVal. A and B are pre-converted bf16 (row-major).
// Block tile 64(M) x 128(N), BK=32, 256 threads = 8 waves (2x4), each wave
// computes a 32x32 tile as 2x2 v_wmma_f32_16x16x32_bf16. Pure-copy staging
// into padded LDS (stride 34 -> conflict-free), prefetch of next K-tile.
// ---------------------------------------------------------------------------
__global__ __launch_bounds__(256)
void gemm_bf16_wmma(const u16* __restrict__ A, const u16* __restrict__ B,
                    const float* __restrict__ bias, float* __restrict__ Cout,
                    int Mrows, int Kd, int Ncols, int scaleCols, float scaleVal)
{
  __shared__ u16 sA[64 * 34];    // [row][k], stride 34
  __shared__ u16 sB[128 * 34];   // transposed: [col][k], stride 34

  const int tid  = threadIdx.x;
  const int gm0  = blockIdx.x * 64;
  const int gn0  = blockIdx.y * 128;
  const int wave = tid >> 5;
  const int lane = tid & 31;
  const int wm   = wave >> 2;               // 0..1
  const int wn   = wave & 3;                // 0..3
  const int half = lane >> 4;               // 0..1
  const int r16  = lane & 15;               // 0..15

  v8f acc[2][2];
#pragma unroll
  for (int i = 0; i < 2; ++i)
#pragma unroll
    for (int j = 0; j < 2; ++j)
#pragma unroll
      for (int e = 0; e < 8; ++e) acc[i][j][e] = 0.0f;

  const int arow = tid >> 2;                // 0..63
  const int ac0  = (tid & 3) * 8;           // 0,8,16,24
  const int bk   = tid & 31;                // 0..31
  const int bc0  = (tid >> 5) * 16;         // 0..112
  const int gr   = gm0 + arow;

  for (int k0 = 0; k0 < Kd; k0 += 32) {
    // --- prefetch next K-tile (global_prefetch_b8; speculative, safe) ---
    if (k0 + 32 < Kd) {
      if (gr < Mrows)
        __builtin_prefetch(A + (size_t)gr * Kd + k0 + 32 + ac0, 0, 0);
      __builtin_prefetch(B + (size_t)(k0 + 32 + bk) * Ncols + gn0 + bc0, 0, 0);
    }
    // --- stage A tile (64x32 bf16): one b128 load, 4 b32 LDS stores ---
    {
      unsigned av[4] = {0u, 0u, 0u, 0u};
      if (gr < Mrows) {
        const uint4 t = *(const uint4*)(A + (size_t)gr * Kd + k0 + ac0);
        av[0] = t.x; av[1] = t.y; av[2] = t.z; av[3] = t.w;
      }
      unsigned* sA32 = (unsigned*)sA;
      const int base = (arow * 34 + ac0) >> 1;   // even -> exact uint index
#pragma unroll
      for (int i = 0; i < 4; ++i) sA32[base + i] = av[i];
    }
    // --- stage B tile (32x128 bf16), stored transposed [col][k] ---
    {
      union { uint4 q; u16 s[8]; } b0, b1;
      const u16* srcB = B + (size_t)(k0 + bk) * Ncols + gn0 + bc0;
      b0.q = *(const uint4*)srcB;
      b1.q = *(const uint4*)(srcB + 8);
#pragma unroll
      for (int i = 0; i < 8; ++i) sB[(bc0 + i) * 34 + bk] = b0.s[i];
#pragma unroll
      for (int i = 0; i < 8; ++i) sB[(bc0 + 8 + i) * 34 + bk] = b1.s[i];
    }
    __syncthreads();

    // --- build fragments per ISA 7.12.2 layout, then WMMA ---
    FragAB af[2], bfr[2];
    const unsigned* pA = (const unsigned*)sA;
    const unsigned* pB = (const unsigned*)sB;
#pragma unroll
    for (int v = 0; v < 8; ++v) {
      const int kk = ((v < 4) ? 2 * v : 16 + 2 * (v - 4)) + 8 * half;
      af[0].u[v]  = pA[((wm * 32 +      r16) * 34 + kk) >> 1];
      af[1].u[v]  = pA[((wm * 32 + 16 + r16) * 34 + kk) >> 1];
      bfr[0].u[v] = pB[((wn * 32 +      r16) * 34 + kk) >> 1];
      bfr[1].u[v] = pB[((wn * 32 + 16 + r16) * 34 + kk) >> 1];
    }
#pragma unroll
    for (int mt = 0; mt < 2; ++mt)
#pragma unroll
      for (int nt = 0; nt < 2; ++nt)
        acc[mt][nt] = __builtin_amdgcn_wmma_f32_16x16x32_bf16(
            false, af[mt].v, false, bfr[nt].v, (short)0, acc[mt][nt],
            false, false);
    __syncthreads();
  }

  // --- epilogue: bias + optional q-scale ---
#pragma unroll
  for (int mt = 0; mt < 2; ++mt) {
#pragma unroll
    for (int nt = 0; nt < 2; ++nt) {
      const int col = gn0 + wn * 32 + nt * 16 + r16;
      if (col >= Ncols) continue;
      const float bcol = bias[col];
      const float sc = (col < scaleCols) ? scaleVal : 1.0f;
#pragma unroll
      for (int e = 0; e < 8; ++e) {
        const int grow = gm0 + wm * 32 + mt * 16 + e + 8 * half;
        if (grow < Mrows)
          Cout[(size_t)grow * Ncols + col] = (acc[mt][nt][e] + bcol) * sc;
      }
    }
  }
}

// ---------------------------------------------------------------------------
// Per-edge relative-position bin indices (cart + sphere), packed 3x6 bits.
// ---------------------------------------------------------------------------
__device__ __forceinline__ void cart2sphere(float x, float y, float z,
                                            float& th, float& be, float& r) {
  const float c = 180.0f / PI_F;
  th = (atan2f(y, x) + PI_F) * c;
  be = atan2f(sqrtf(x * x + y * y), z) * c;
  r  = sqrtf(x * x + y * y + z * z);
}

__global__ __launch_bounds__(256)
void relidx_kernel(const float* __restrict__ xyz,
                   const int* __restrict__ i0c, const int* __restrict__ i1c,
                   const int* __restrict__ i0s, const int* __restrict__ i1s,
                   unsigned* __restrict__ relc, unsigned* __restrict__ rels)
{
  const int m = blockIdx.x * blockDim.x + threadIdx.x;
  if (m >= M_EDG) return;

  // --- cartesian branch: idx = clip(floor(rel/0.25)+23, 0, 46) ---
  {
    const int a = i0c[m], b = i1c[m];
    unsigned pk = 0;
#pragma unroll
    for (int d = 0; d < 3; ++d) {
      const float rel = xyz[a * 3 + d] - xyz[b * 3 + d];
      int idx = (int)floorf(rel * 4.0f) + 23;
      idx = min(max(idx, 0), 46);
      pk |= (unsigned)idx << (6 * d);
    }
    relc[m] = pk;
  }
  // --- sphere branch ---
  {
    const int a = i0s[m], b = i1s[m];
    float ta, ba, ra, tb, bb, rb;
    cart2sphere(xyz[a * 3], xyz[a * 3 + 1], xyz[a * 3 + 2], ta, ba, ra);
    cart2sphere(xyz[b * 3], xyz[b * 3 + 1], xyz[b * 3 + 2], tb, bb, rb);
    const float r0 = ta - tb, r1 = ba - bb, r2 = ra - rb;
    int j0 = min(max((int)floorf(r0 * 0.2f) + 24, 0), 47);
    int j1 = min(max((int)floorf(r1 * 0.2f) + 24, 0), 47);
    const float rabs = fabsf(r2);
    const float flag = (r2 >= 0.0f) ? 1.0f : 0.0f;
    float fi = 2.0f * floorf(log2f((rabs + 2.0f * A_CONST) / A_CONST)) - 2.0f;
    fi += ((3.0f * exp2f(floorf(fi * 0.5f)) - 2.0f) * A_CONST <= rabs) ? 1.0f : 0.0f;
    fi = fi * (2.0f * flag - 1.0f) + (flag - 1.0f);
    int j2 = min(max((int)fi + 24, 0), 47);
    rels[m] = (unsigned)j0 | ((unsigned)j1 << 6) | ((unsigned)j2 << 12);
  }
}

// ---------------------------------------------------------------------------
// Gather attention branch. One wave per (point, head). Segments are the 16
// contiguous edges of each point, so softmax is purely in-register.
// Each lane owns 2 of the 64 head dims. Output written directly as bf16
// (packed pair per lane) so the proj GEMM needs no conversion pass.
// ---------------------------------------------------------------------------
__device__ __forceinline__ float2 tab2(const float* __restrict__ T,
                                       int e, int dim, int h, int d0) {
  return *(const float2*)(T + ((((size_t)e * 3 + dim) * 3 + h) * HD_) + d0);
}

__global__ __launch_bounds__(256)
void attn_branch(const float* __restrict__ qkv,        // N x 1152 (q|k|v)
                 const int* __restrict__ idx1,         // M
                 const unsigned* __restrict__ relp,    // M packed
                 const float* __restrict__ Tq, const float* __restrict__ Tk,
                 const float* __restrict__ Tv,
                 unsigned* __restrict__ xout_bf,       // N x 384 bf16 (as u32 pairs)
                 int h0)
{
  const int w    = (blockIdx.x * blockDim.x + threadIdx.x) >> 5;
  const int lane = threadIdx.x & 31;
  if (w >= N_PTS * H1_) return;
  const int n  = w / H1_;
  const int h  = w - n * H1_;     // local head 0..2 (table index)
  const int hg = h0 + h;          // global head 0..5
  const int d0 = lane * 2;

  const float2 qv = *(const float2*)(qkv + (size_t)n * 1152 + hg * HD_ + d0);

  float    att[KNBR];
  int      nb[KNBR];
  unsigned pk[KNBR];

#pragma unroll
  for (int j = 0; j < KNBR; ++j) {
    const int m = n * KNBR + j;
    const int i1 = idx1[m];
    const unsigned p = relp[m];
    nb[j] = i1; pk[j] = p;
    const int e0 = p & 63, e1 = (p >> 6) & 63, e2 = (p >> 12) & 63;

    const float2 kg = *(const float2*)(qkv + (size_t)i1 * 1152 + CCH + hg * HD_ + d0);
    const float2 tq0 = tab2(Tq, e0, 0, h, d0);
    const float2 tq1 = tab2(Tq, e1, 1, h, d0);
    const float2 tq2 = tab2(Tq, e2, 2, h, d0);
    const float2 tk0 = tab2(Tk, e0, 0, h, d0);
    const float2 tk1 = tab2(Tk, e1, 1, h, d0);
    const float2 tk2 = tab2(Tk, e2, 2, h, d0);
    const float tqsx = tq0.x + tq1.x + tq2.x, tqsy = tq0.y + tq1.y + tq2.y;
    const float tksx = tk0.x + tk1.x + tk2.x, tksy = tk0.y + tk1.y + tk2.y;

    float t = qv.x * (kg.x + tqsx) + qv.y * (kg.y + tqsy)
            + kg.x * tksx + kg.y * tksy;
#pragma unroll
    for (int o = 1; o < 32; o <<= 1) t += __shfl_xor(t, o, 32);
    att[j] = t;
  }

  // in-register softmax over the 16 neighbors
  float mx = att[0];
#pragma unroll
  for (int j = 1; j < KNBR; ++j) mx = fmaxf(mx, att[j]);
  float s = 0.0f;
#pragma unroll
  for (int j = 0; j < KNBR; ++j) { att[j] = __expf(att[j] - mx); s += att[j]; }
  const float inv = 1.0f / s;

  float accx = 0.0f, accy = 0.0f;
#pragma unroll
  for (int j = 0; j < KNBR; ++j) {
    const float p = att[j] * inv;
    const unsigned pj = pk[j];
    const int e0 = pj & 63, e1 = (pj >> 6) & 63, e2 = (pj >> 12) & 63;
    const float2 vg = *(const float2*)(qkv + (size_t)nb[j] * 1152 + 2 * CCH + hg * HD_ + d0);
    const float2 tv0 = tab2(Tv, e0, 0, h, d0);
    const float2 tv1 = tab2(Tv, e1, 1, h, d0);
    const float2 tv2 = tab2(Tv, e2, 2, h, d0);
    accx += p * (vg.x + tv0.x + tv1.x + tv2.x);
    accy += p * (vg.y + tv0.y + tv1.y + tv2.y);
  }
  // write packed bf16 pair: x[n][hg*64 + d0 .. d0+1]
  xout_bf[(size_t)n * (CCH / 2) + hg * (HD_ / 2) + lane] =
      (unsigned)f2bf(accx) | ((unsigned)f2bf(accy) << 16);
}

// ---------------------------------------------------------------------------
extern "C" void kernel_launch(void* const* d_in, const int* in_sizes, int n_in,
                              void* d_out, int out_size, void* d_ws, size_t ws_size,
                              hipStream_t stream) {
  const float* query_feats = (const float*)d_in[0];
  const float* xyz         = (const float*)d_in[1];
  const int*   i0c         = (const int*)d_in[2];
  const int*   i1c         = (const int*)d_in[3];
  const int*   i0s         = (const int*)d_in[4];
  const int*   i1s         = (const int*)d_in[5];
  const float* W_qkv       = (const float*)d_in[6];
  const float* b_qkv       = (const float*)d_in[7];
  const float* W_proj      = (const float*)d_in[8];
  const float* b_proj      = (const float*)d_in[9];
  const float* tq          = (const float*)d_in[10];
  const float* tk          = (const float*)d_in[11];
  const float* tv          = (const float*)d_in[12];
  const float* tq_s        = (const float*)d_in[13];
  const float* tk_s        = (const float*)d_in[14];
  const float* tv_s        = (const float*)d_in[15];
  float* out = (float*)d_out;

  char* ws = (char*)d_ws;
  size_t off = 0;
  float* qkv = (float*)(ws + off);          off += (size_t)N_PTS * 1152 * sizeof(float);
  unsigned* relc = (unsigned*)(ws + off);   off += (size_t)M_EDG * sizeof(unsigned);
  unsigned* rels = (unsigned*)(ws + off);   off += (size_t)M_EDG * sizeof(unsigned);
  u16* qf_bf    = (u16*)(ws + off);         off += (size_t)N_PTS * CCH * sizeof(u16);
  u16* xbuf_bf  = (u16*)(ws + off);         off += (size_t)N_PTS * CCH * sizeof(u16);
  u16* wqkv_bf  = (u16*)(ws + off);         off += (size_t)CCH * 3 * CCH * sizeof(u16);
  u16* wproj_bf = (u16*)(ws + off);         off += (size_t)CCH * CCH * sizeof(u16);

  const float SCALE = 0.125f;   // 64^-0.5

  // 0) one-time fp32 -> bf16 conversions
  {
    const int p_qf = N_PTS * CCH / 2;
    const int p_wq = CCH * 3 * CCH / 2;
    const int p_wp = CCH * CCH / 2;
    f32_to_bf16_kernel<<<(p_qf + 255) / 256, 256, 0, stream>>>(query_feats, (unsigned*)qf_bf, p_qf);
    f32_to_bf16_kernel<<<(p_wq + 255) / 256, 256, 0, stream>>>(W_qkv, (unsigned*)wqkv_bf, p_wq);
    f32_to_bf16_kernel<<<(p_wp + 255) / 256, 256, 0, stream>>>(W_proj, (unsigned*)wproj_bf, p_wp);
  }
  // 1) fused QKV GEMM (+bias, q pre-scaled)
  {
    dim3 grid((N_PTS + 63) / 64, (3 * CCH) / 128);
    gemm_bf16_wmma<<<grid, 256, 0, stream>>>(qf_bf, wqkv_bf, b_qkv, qkv,
                                             N_PTS, CCH, 3 * CCH, CCH, SCALE);
  }
  // 2) relative-position bin indices (both branches)
  relidx_kernel<<<(M_EDG + 255) / 256, 256, 0, stream>>>(xyz, i0c, i1c, i0s, i1s,
                                                         relc, rels);
  // 3) attention branches (one wave per point-head), write x as bf16
  {
    const int nwaves = N_PTS * H1_;
    const int blocks = (nwaves + 7) / 8;      // 8 waves / 256-thread block
    attn_branch<<<blocks, 256, 0, stream>>>(qkv, i1c, relc, tq, tk, tv,
                                            (unsigned*)xbuf_bf, 0);
    attn_branch<<<blocks, 256, 0, stream>>>(qkv, i1s, rels, tq_s, tk_s, tv_s,
                                            (unsigned*)xbuf_bf, H1_);
  }
  // 4) output projection
  {
    dim3 grid((N_PTS + 63) / 64, CCH / 128);
    gemm_bf16_wmma<<<grid, 256, 0, stream>>>(xbuf_bf, wproj_bf, b_proj, out,
                                             N_PTS, CCH, CCH, 0, 1.0f);
  }
}
